// GINCodeModel_90202903150610
// MI455X (gfx1250) — compile-verified, matrix operand order checked by reading, not run on hardware
//
#include <hip/hip_runtime.h>
#include <math.h>

// ---------------------------------------------------------------------------
// GIN (Graph Isomorphism Network) forward for MI455X / gfx1250.
// D = 256 hidden, L = 3 GIN layers, classifier head, sigmoid output.
//
// Input order (setup_inputs dict order, nested structures flattened in
// insertion order):
//   0: emb [V*D] f32
//   1..4  : layer0 w1[D*D], b1[D], w2[D*D], b2[D]
//   5..8  : layer1 w1, b1, w2, b2
//   9..12 : layer2 w1, b1, w2, b2
//   13: eps [3] f32
//   14: cls_w1 [D*D]   15: cls_b1 [D]   16: cls_w2 [D]   17: cls_b2 [1]
//   18: x [N] int32    19: edge_index [2*E] int32 (src then dst)
//   20: batch [N] int32
// Output: [G] f32 (sigmoid scores).
// ---------------------------------------------------------------------------

#define DIM 256
#define DIM4 64            // DIM / 4
#define NT  4              // C-tiles (16 cols each) per wave: 16x64 wave tile

typedef __attribute__((ext_vector_type(16))) _Float16 v16h;
typedef __attribute__((ext_vector_type(8)))  _Float16 v8h;
typedef __attribute__((ext_vector_type(4)))  _Float16 h4;
typedef __attribute__((ext_vector_type(8)))  float    v8f;

// ---------------- elementwise / memory kernels ----------------

__global__ void k_zero_f4(float* __restrict__ p, int n4) {
  int tid = blockIdx.x * blockDim.x + threadIdx.x;
  if (tid < n4) reinterpret_cast<float4*>(p)[tid] = make_float4(0.f, 0.f, 0.f, 0.f);
}

// h[n, :] = emb[x[n], :]
__global__ void k_embed(const float* __restrict__ emb, const int* __restrict__ x,
                        float* __restrict__ h, int n_nodes) {
  int tid = blockIdx.x * blockDim.x + threadIdx.x;
  if (tid >= n_nodes * DIM4) return;
  int n = tid >> 6;
  int c = tid & 63;
  int row = x[n];
  reinterpret_cast<float4*>(h)[n * DIM4 + c] =
      reinterpret_cast<const float4*>(emb)[(long)row * DIM4 + c];
}

// agg[dst[e], :] += h[src[e], :]   (L2-resident float atomics)
__global__ void k_scatter_add(const float* __restrict__ h, const int* __restrict__ src,
                              const int* __restrict__ dst, float* __restrict__ agg,
                              int n_edges) {
  int tid = blockIdx.x * blockDim.x + threadIdx.x;
  if (tid >= n_edges * DIM4) return;
  int e = tid >> 6;
  int c = tid & 63;
  int s = src[e];
  int d = dst[e];
  float4 v = reinterpret_cast<const float4*>(h)[s * DIM4 + c];
  float* out = agg + d * DIM + c * 4;
  atomicAdd(out + 0, v.x);
  atomicAdd(out + 1, v.y);
  atomicAdd(out + 2, v.z);
  atomicAdd(out + 3, v.w);
}

// u16 = f16( (1+eps[layer]) * h + agg )
__global__ void k_pre_f16(const float* __restrict__ h, const float* __restrict__ agg,
                          const float* __restrict__ eps, int layer,
                          _Float16* __restrict__ u, int n_nodes) {
  int tid = blockIdx.x * blockDim.x + threadIdx.x;
  if (tid >= n_nodes * DIM4) return;
  float e1 = 1.0f + eps[layer];
  float4 a = reinterpret_cast<const float4*>(h)[tid];
  float4 b = reinterpret_cast<const float4*>(agg)[tid];
  h4 r;
  r[0] = (_Float16)(e1 * a.x + b.x);
  r[1] = (_Float16)(e1 * a.y + b.y);
  r[2] = (_Float16)(e1 * a.z + b.z);
  r[3] = (_Float16)(e1 * a.w + b.w);
  reinterpret_cast<h4*>(u)[tid] = r;
}

// wt[j, k] = f16( w[k, j] )  -- transpose so WMMA B-fragments are contiguous
__global__ void k_wcvt_t(const float* __restrict__ w, _Float16* __restrict__ wt) {
  int tid = blockIdx.x * blockDim.x + threadIdx.x;   // DIM*DIM threads
  int k = tid >> 8;
  int j = tid & 255;
  wt[j * DIM + k] = (_Float16)w[k * DIM + j];
}

// hg[batch[n], :] += h[n, :]
__global__ void k_pool(const float* __restrict__ h, const int* __restrict__ batch,
                       float* __restrict__ hg, int n_nodes) {
  int tid = blockIdx.x * blockDim.x + threadIdx.x;
  if (tid >= n_nodes * DIM4) return;
  int n = tid >> 6;
  int c = tid & 63;
  int g = batch[n];
  float4 v = reinterpret_cast<const float4*>(h)[n * DIM4 + c];
  float* out = hg + g * DIM + c * 4;
  atomicAdd(out + 0, v.x);
  atomicAdd(out + 1, v.y);
  atomicAdd(out + 2, v.z);
  atomicAdd(out + 3, v.w);
}

// ---------------- WMMA GEMM: C = relu(A[M,256] * Bt^T + bias) ----------------
// A: f16 row-major [M,256].  Bt: f16 [256(out), 256(k)] (transposed weights).
// Writes f16 (C16) and/or f32 (C32) row-major [M,256]; null pointer disables.
//
// Fragment addressing per CDNA5 ISA 7.12.2 (16-bit A 16x32): lane l holds
// row (l&15); its 16 halves are K = 8*(l>>4)+0..7 then +16 — two contiguous
// 16-byte chunks -> two global_load_b128 per fragment.
//
// Block = 256 threads = 8 waves arranged 2(M) x 4(N): 32 rows x 256 cols per
// block. Wave tile 16x64 (4 accum tiles). K-loop is explicitly double-
// buffered: fragments for step k+1 are issued before the WMMAs of step k so
// s_wait_loadcnt never drains to zero mid-stream.

__device__ __forceinline__ v16h load_frag16(const _Float16* __restrict__ p) {
  v8h lo = *reinterpret_cast<const v8h*>(p);
  v8h hi = *reinterpret_cast<const v8h*>(p + 16);
  v16h f;
#pragma unroll
  for (int i = 0; i < 8; ++i) { f[i] = lo[i]; f[i + 8] = hi[i]; }
  return f;
}

__global__ __launch_bounds__(256)
void k_gemm_bias_relu(const _Float16* __restrict__ A, const _Float16* __restrict__ Bt,
                      const float* __restrict__ bias,
                      _Float16* __restrict__ C16, float* __restrict__ C32, int M) {
  const int lane = threadIdx.x & 31;
  const int wave = threadIdx.x >> 5;
  const int wm = wave >> 2;                 // 0..1 : which 16-row strip
  const int wn = wave & 3;                  // 0..3 : which 64-col quarter
  const int m0 = blockIdx.x * 32 + wm * 16;
  const int n0 = wn * 64;
  const int lm = lane & 15;                 // row (A) / col (B) within tile
  const int lh = lane >> 4;                 // half selector

  int arow = m0 + lm;
  if (arow >= M) arow = M - 1;              // clamp (stores are masked)
  const _Float16* aptr = A + (long)arow * DIM + lh * 8;
  const _Float16* bbase = Bt + (long)(n0 + lm) * DIM + lh * 8;

  v8f acc[NT];
#pragma unroll
  for (int t = 0; t < NT; ++t)
#pragma unroll
    for (int i = 0; i < 8; ++i) acc[t][i] = 0.0f;

  // prologue: fragments for k0 = 0
  v16h a = load_frag16(aptr);
  v16h b[NT];
#pragma unroll
  for (int t = 0; t < NT; ++t) b[t] = load_frag16(bbase + (long)t * 16 * DIM);

#pragma unroll
  for (int k0 = 0; k0 < DIM; k0 += 32) {
    v16h an;
    v16h bn[NT];
    const bool last = (k0 + 32 >= DIM);
    if (!last) {
      // issue next step's loads before consuming current fragments
      __builtin_prefetch(aptr + k0 + 64, 0, 1);
      an = load_frag16(aptr + k0 + 32);
#pragma unroll
      for (int t = 0; t < NT; ++t)
        bn[t] = load_frag16(bbase + (long)t * 16 * DIM + k0 + 32);
    }
#pragma unroll
    for (int t = 0; t < NT; ++t) {
      acc[t] = __builtin_amdgcn_wmma_f32_16x16x32_f16(
          /*neg_a=*/false, a, /*neg_b=*/false, b[t],
          /*c_mod=*/(short)0, acc[t], /*reuse_a=*/false, /*reuse_b=*/false);
    }
    if (!last) {
      a = an;
#pragma unroll
      for (int t = 0; t < NT; ++t) b[t] = bn[t];
    }
  }

  // epilogue: bias + relu, masked stores
#pragma unroll
  for (int t = 0; t < NT; ++t) {
    int col = n0 + t * 16 + lm;
    float bv = bias[col];
#pragma unroll
    for (int r = 0; r < 8; ++r) {
      int row = m0 + r + 8 * lh;
      if (row < M) {
        float v = acc[t][r] + bv;
        v = v > 0.f ? v : 0.f;
        if (C32) C32[(long)row * DIM + col] = v;
        if (C16) C16[(long)row * DIM + col] = (_Float16)v;
      }
    }
  }
}

// ---------------- classifier head (tiny: 128 x [256x256] + [256x1]) ----------

__global__ void k_classifier(const float* __restrict__ hg, const float* __restrict__ w1,
                             const float* __restrict__ b1, const float* __restrict__ w2,
                             const float* __restrict__ b2, float* __restrict__ out) {
  __shared__ float xs[DIM];
  __shared__ float red[DIM];
  int g = blockIdx.x, t = threadIdx.x;
  xs[t] = hg[g * DIM + t];
  __syncthreads();
  float s = 0.f;
#pragma unroll 8
  for (int k = 0; k < DIM; ++k) s += xs[k] * w1[k * DIM + t];
  s += b1[t];
  s = s > 0.f ? s : 0.f;
  red[t] = s * w2[t];
  __syncthreads();
  for (int off = DIM / 2; off > 0; off >>= 1) {
    if (t < off) red[t] += red[t + off];
    __syncthreads();
  }
  if (t == 0) {
    float z = red[0] + b2[0];
    out[g] = 1.0f / (1.0f + __expf(-z));
  }
}

// ---------------- host side ----------------

static inline size_t align256(size_t v) { return (v + 255) & ~(size_t)255; }

extern "C" void kernel_launch(void* const* d_in, const int* in_sizes, int n_in,
                              void* d_out, int out_size, void* d_ws, size_t ws_size,
                              hipStream_t stream) {
  (void)n_in; (void)ws_size;
  const float* emb    = (const float*)d_in[0];
  const float* w1[3]  = {(const float*)d_in[1], (const float*)d_in[5], (const float*)d_in[9]};
  const float* b1[3]  = {(const float*)d_in[2], (const float*)d_in[6], (const float*)d_in[10]};
  const float* w2[3]  = {(const float*)d_in[3], (const float*)d_in[7], (const float*)d_in[11]};
  const float* b2[3]  = {(const float*)d_in[4], (const float*)d_in[8], (const float*)d_in[12]};
  const float* eps    = (const float*)d_in[13];
  const float* cls_w1 = (const float*)d_in[14];
  const float* cls_b1 = (const float*)d_in[15];
  const float* cls_w2 = (const float*)d_in[16];
  const float* cls_b2 = (const float*)d_in[17];
  const int*   x      = (const int*)d_in[18];
  const int*   edge   = (const int*)d_in[19];
  const int*   batch  = (const int*)d_in[20];

  const int N_ = in_sizes[18];          // nodes (20000)
  const int E_ = in_sizes[19] / 2;      // edges (320000)
  const int G_ = out_size;              // graphs (128)
  const int* src = edge;
  const int* dst = edge + E_;

  // workspace carve-up
  char* p = (char*)d_ws;
  float*    h    = (float*)p;    p += align256((size_t)N_ * DIM * sizeof(float));
  float*    agg  = (float*)p;    p += align256((size_t)N_ * DIM * sizeof(float));
  _Float16* u16  = (_Float16*)p; p += align256((size_t)N_ * DIM * sizeof(_Float16));
  _Float16* t16  = (_Float16*)p; p += align256((size_t)N_ * DIM * sizeof(_Float16));
  _Float16* w1t  = (_Float16*)p; p += align256((size_t)DIM * DIM * sizeof(_Float16));
  _Float16* w2t  = (_Float16*)p; p += align256((size_t)DIM * DIM * sizeof(_Float16));
  float*    hg   = (float*)p;    p += align256((size_t)G_ * DIM * sizeof(float));

  const int THR = 256;
  const int nd4   = N_ * DIM4;                         // node-level float4 work items
  const int nd4_g = (nd4 + THR - 1) / THR;
  const int ed4_g = (E_ * DIM4 + THR - 1) / THR;
  const int gemm_g = (N_ + 31) / 32;                   // 32 rows per block

  // h = emb[x]
  k_embed<<<nd4_g, THR, 0, stream>>>(emb, x, h, N_);

  for (int l = 0; l < 3; ++l) {
    // convert + transpose this layer's weights to f16
    k_wcvt_t<<<DIM, DIM, 0, stream>>>(w1[l], w1t);
    k_wcvt_t<<<DIM, DIM, 0, stream>>>(w2[l], w2t);
    // agg = segment_sum(h[src], dst)
    k_zero_f4<<<nd4_g, THR, 0, stream>>>(agg, nd4);
    k_scatter_add<<<ed4_g, THR, 0, stream>>>(h, src, dst, agg, E_);
    // u = f16((1+eps)*h + agg)
    k_pre_f16<<<nd4_g, THR, 0, stream>>>(h, agg, eps, l, u16, N_);
    // t = relu(u @ w1 + b1)  (f16 out, feeds next WMMA)
    k_gemm_bias_relu<<<gemm_g, THR, 0, stream>>>(u16, w1t, b1[l], t16, nullptr, N_);
    // h = relu(t @ w2 + b2)  (f32 out, feeds scatter/pool)
    k_gemm_bias_relu<<<gemm_g, THR, 0, stream>>>(t16, w2t, b2[l], nullptr, h, N_);
  }

  // global add pool
  k_zero_f4<<<(G_ * DIM4 + THR - 1) / THR, THR, 0, stream>>>(hg, G_ * DIM4);
  k_pool<<<nd4_g, THR, 0, stream>>>(h, batch, hg, N_);

  // classifier + sigmoid
  k_classifier<<<G_, DIM, 0, stream>>>(hg, cls_w1, cls_b1, cls_w2, cls_b2, (float*)d_out);
}
